// AttentionSubsample_30202210025656
// MI455X (gfx1250) — matrix operand order repaired and checked by reading, not compiled, not run on hardware
//
#include <hip/hip_runtime.h>
#include <hip/hip_bf16.h>
#include <cstdint>
#include <cstddef>

// ---- problem constants (match reference) ----
#define B_    128
#define NTOK  785      // N = 28*28+1
#define PP    784
#define CC    256
#define HH    8
#define KDIM  16
#define DV    32
#define DH_   256      // H*DV
#define HTOT_ 384
#define NHKD_ 128
#define OUTC  384
#define QQ    197      // 14*14+1
#define QT    13       // ceil(197/16)
#define QPAD  208      // QT*16
#define NPAD  800      // 785 padded to mult of 32 (and of 16)
#define NT50  50       // NPAD/16 m-tiles per batch in kv GEMM
#define EPSF  1e-5f
#define SCALEF 0.25f   // KD^-0.5

typedef _Float16 half_t;
typedef __attribute__((ext_vector_type(16))) _Float16 v16h;
typedef __attribute__((ext_vector_type(8)))  float    v8f;

// 16-bit A-matrix 16x32 per-lane element->K mapping (ISA 7.12.2)
__device__ __forceinline__ int a_frag_k(int e, int hf) {
  int v = e >> 1;
  int base = (v < 4) ? (2 * v) : (8 + 2 * v);
  return base + 8 * hf + (e & 1);
}

__device__ __forceinline__ v8f wmma_f16(v16h a, v16h b, v8f c) {
  return __builtin_amdgcn_wmma_f32_16x16x32_f16(false, a, false, b, (short)0, c, false, false);
}

__device__ __forceinline__ int q_rowmap(int qg) {
  // row in x_full (within one batch) for query index qg (subsample_cls mapping)
  if (qg <= 0 || qg >= QQ) return 0;
  int t = qg - 1;
  int i = t / 14, j = t % 14;
  return 1 + i * 56 + j * 2;
}

// ---------- K0: inverse permutation + old_scale ----------
__global__ __launch_bounds__(256)
void perm_oldscale(const float* __restrict__ ori, const float* __restrict__ ga,
                   int* __restrict__ idxbuf, float* __restrict__ oldscale) {
  int b = blockIdx.x, tid = threadIdx.x;
  float s = 0.f;
  for (int p = tid; p < PP; p += 256) {
    int r = (int)(ori[(size_t)b * PP + p] + 0.5f);   // ori values are exact ranks
    idxbuf[(size_t)b * PP + r] = p;                   // idx = argsort(ori)
    s += ga[(size_t)b * PP + p];
  }
  __shared__ float red[256];
  red[tid] = s; __syncthreads();
  for (int o = 128; o > 0; o >>= 1) { if (tid < o) red[tid] += red[tid + o]; __syncthreads(); }
  if (tid == 0) oldscale[b] = red[0];
}

// ---------- K1: fold BN into scale/bias, convert weights to f16 ----------
__global__ __launch_bounds__(256)
void fold_weights(const float* __restrict__ kv_w, const float* __restrict__ q_w, const float* __restrict__ pj_w,
                  const float* kv_g, const float* kv_b, const float* kv_m, const float* kv_v,
                  const float* q_g,  const float* q_b,  const float* q_m,  const float* q_v,
                  const float* p_g,  const float* p_b,  const float* p_m,  const float* p_v,
                  half_t* __restrict__ kvw_h, half_t* __restrict__ qw_h, half_t* __restrict__ pjw_h,
                  float* kv_s, float* kv_t, float* q_s, float* q_t, float* pj_s, float* pj_t) {
  int i = blockIdx.x * 256 + threadIdx.x;
  if (i < HTOT_ * CC) kvw_h[i] = (half_t)kv_w[i];
  if (i < NHKD_ * CC) qw_h[i]  = (half_t)q_w[i];
  if (i < OUTC  * CC) pjw_h[i] = (half_t)pj_w[i];
  if (i < HTOT_) { float sc = kv_g[i] * rsqrtf(kv_v[i] + EPSF); kv_s[i] = sc; kv_t[i] = kv_b[i] - kv_m[i] * sc; }
  if (i < NHKD_) { float sc = q_g[i]  * rsqrtf(q_v[i]  + EPSF); q_s[i] = sc * SCALEF; q_t[i] = (q_b[i] - q_m[i] * sc) * SCALEF; }
  if (i < OUTC)  { float sc = p_g[i]  * rsqrtf(p_v[i]  + EPSF); pj_s[i] = sc; pj_t[i] = p_b[i] - p_m[i] * sc; }
}

// ---------- K2: gather by inverse permutation, add pos, convert to f16 ----------
// x_h is (B, NPAD, CC); rows >= NTOK are zeroed (so kv GEMM pad tiles are finite).
__global__ __launch_bounds__(256)
void gather_pos(const float* __restrict__ x, const float* __restrict__ poss,
                const int* __restrict__ idxbuf, half_t* __restrict__ xh) {
  int b = blockIdx.x / NPAD;
  int n = blockIdx.x % NPAD;
  int c = threadIdx.x;
  float val;
  if (n >= NTOK) {
    val = 0.f;
  } else {
    size_t srcrow;
    if (n == 0) srcrow = (size_t)b * NTOK;
    else        srcrow = (size_t)b * NTOK + 1 + idxbuf[(size_t)b * PP + (n - 1)];
    val = x[srcrow * CC + c] + poss[(size_t)n * CC + c];
  }
  xh[((size_t)b * NPAD + n) * CC + c] = (half_t)val;
}

// ---------- K3: KV GEMM per batch (NPAD x 256) @ (256 x 384), BN fold, split K/V ----------
// Pad rows (n>=785) get the BN bias -> finite; softmax masks them to prob 0.
__global__ __launch_bounds__(256)
void kv_gemm(const half_t* __restrict__ Xh, const half_t* __restrict__ Wh,
             const float* __restrict__ s, const float* __restrict__ t,
             half_t* __restrict__ Kb, half_t* __restrict__ Vb) {
  int wid  = (blockIdx.x * 256 + threadIdx.x) >> 5;
  int lane = threadIdx.x & 31;
  int b  = wid / (NT50 * 24);
  int rem = wid % (NT50 * 24);
  int mt = rem / 24, nt = rem % 24;     // 50 m-tiles x 24 n-tiles per batch
  int n0 = nt << 4;
  int mr = lane & 15, hf = lane >> 4;
  const half_t* arow = Xh + ((size_t)b * NPAD + (mt << 4) + mr) * CC;
  const half_t* brow = Wh + (size_t)(n0 + mr) * CC;    // W row-major (out,ch); B[k][n]=W[n][k]
  v8f acc = {};
  for (int k0 = 0; k0 < CC; k0 += 32) {
    v16h a, bfr;
#pragma unroll
    for (int e = 0; e < 16; ++e) {
      a[e]   = arow[k0 + a_frag_k(e, hf)];
      bfr[e] = brow[k0 + a_frag_k(e, hf)];
    }
    acc = wmma_f16(a, bfr, acc);
  }
  int col = n0 + (lane & 15);
  float sc = s[col], tc = t[col];
  int h = col / 48, cc2 = col % 48;
  size_t bh = (size_t)b * HH + h;
#pragma unroll
  for (int r = 0; r < 8; ++r) {
    int nn = (mt << 4) + r + 8 * hf;     // token row within batch (0..799), shift/add only
    float y = acc[r] * sc + tc;
    if (cc2 < KDIM) Kb[(bh * NPAD + nn) * KDIM + cc2] = (half_t)y;
    else            Vb[(bh * NPAD + nn) * DV + (cc2 - KDIM)] = (half_t)y;
  }
}

// ---------- K4: Q GEMM on subsampled rows, BN(+SCALE) fold ----------
__global__ __launch_bounds__(256)
void q_gemm(const half_t* __restrict__ Xh, const half_t* __restrict__ Wh,
            const float* __restrict__ s, const float* __restrict__ t,
            half_t* __restrict__ Qb) {
  int wid  = (blockIdx.x * 256 + threadIdx.x) >> 5;
  int lane = threadIdx.x & 31;
  int b = wid / (QT * 8); int rem = wid % (QT * 8);
  int qt = rem / 8, nt = rem % 8;
  int mr = lane & 15, hf = lane >> 4;
  int row = q_rowmap(qt * 16 + mr);
  const half_t* arow = Xh + ((size_t)b * NPAD + row) * CC;
  const half_t* brow = Wh + (size_t)(nt * 16 + mr) * CC;
  v8f acc = {};
  for (int k0 = 0; k0 < CC; k0 += 32) {
    v16h a, bfr;
#pragma unroll
    for (int e = 0; e < 16; ++e) {
      a[e]   = arow[k0 + a_frag_k(e, hf)];
      bfr[e] = brow[k0 + a_frag_k(e, hf)];
    }
    acc = wmma_f16(a, bfr, acc);
  }
  int col = nt * 16 + (lane & 15);
  float sc = s[col], tc = t[col];
  int h = col >> 4, d = col & 15;
#pragma unroll
  for (int r = 0; r < 8; ++r) {
    int qrow = qt * 16 + r + 8 * hf;
    Qb[(((size_t)b * HH + h) * QPAD + qrow) * KDIM + d] = (half_t)(acc[r] * sc + tc);
  }
}

// ---------- K5: fused attention per (b,h,qtile): QK^T -> LDS softmax -> AV -> hswish ----------
__global__ __launch_bounds__(32)
void attn_kernel(const half_t* __restrict__ Qb, const half_t* __restrict__ Kb,
                 const half_t* __restrict__ Vb, float* __restrict__ cls_part,
                 half_t* __restrict__ Ob) {
  __shared__ float S[16 * NPAD];   // 51.2 KB score/prob tile, never touches HBM
  int idx = blockIdx.x;
  int b = idx / (HH * QT); int rem = idx % (HH * QT);
  int h = rem / QT, qt = rem % QT;
  int lane = threadIdx.x;
  int mr = lane & 15, hf = lane >> 4;
  int q0 = qt * 16;
  const half_t* qbase = Qb + (((size_t)b * HH + h) * QPAD + q0) * KDIM;
  const half_t* kbase = Kb + ((size_t)b * HH + h) * NPAD * KDIM;
  const half_t* vbase = Vb + ((size_t)b * HH + h) * NPAD * DV;

  // A fragment: 16 queries x K=32 (d=16 zero-padded)
  v16h aq;
#pragma unroll
  for (int e = 0; e < 16; ++e) {
    int k = a_frag_k(e, hf);
    aq[e] = (k < KDIM) ? qbase[(size_t)mr * KDIM + k] : (half_t)0.f;
  }
  // S = Q K^T (softmax scale folded into Q)
  for (int n0 = 0; n0 < NPAD; n0 += 16) {
    v16h bk;
#pragma unroll
    for (int e = 0; e < 16; ++e) {
      int d = a_frag_k(e, hf);
      bk[e] = (d < KDIM) ? kbase[(size_t)(n0 + mr) * KDIM + d] : (half_t)0.f;
    }
    v8f c = {};
    c = wmma_f16(aq, bk, c);
#pragma unroll
    for (int r = 0; r < 8; ++r) S[(r + 8 * hf) * NPAD + n0 + mr] = c[r];
  }
  __syncthreads();

  // row-wise softmax over 785 valid keys (pad -> prob 0)
  for (int row = 0; row < 16; ++row) {
    float mx = -1e30f;
    for (int c2 = lane; c2 < NPAD; c2 += 32)
      if (c2 < NTOK) mx = fmaxf(mx, S[row * NPAD + c2]);
#pragma unroll
    for (int o = 16; o >= 1; o >>= 1) mx = fmaxf(mx, __shfl_xor(mx, o));
    float sum = 0.f;
    for (int c2 = lane; c2 < NPAD; c2 += 32) {
      float p = (c2 < NTOK) ? __expf(S[row * NPAD + c2] - mx) : 0.f;
      S[row * NPAD + c2] = p;
      sum += p;
    }
#pragma unroll
    for (int o = 16; o >= 1; o >>= 1) sum += __shfl_xor(sum, o);
    float inv = 1.f / sum;
    for (int c2 = lane; c2 < NPAD; c2 += 32) S[row * NPAD + c2] *= inv;
  }
  __syncthreads();

  // cls attention row (q=0) -> per-(b,h) subsampled slice (deterministic, no atomics)
  if (qt == 0) {
    for (int t2 = lane; t2 < 196; t2 += 32) {
      int i = t2 / 14, j = t2 % 14;
      cls_part[((size_t)b * HH + h) * 196 + t2] = S[1 + i * 56 + 2 * j];
    }
  }
  __syncthreads();

  // O = P @ V  (two 16-wide column blocks of dv=32)
  v8f acc0 = {}, acc1 = {};
  for (int kc = 0; kc < NPAD; kc += 32) {
    v16h ap, bv0, bv1;
#pragma unroll
    for (int e = 0; e < 16; ++e) {
      ap[e] = (half_t)S[mr * NPAD + kc + a_frag_k(e, hf)];
      int key = kc + a_frag_k(e, hf);
      bv0[e] = vbase[(size_t)key * DV + mr];
      bv1[e] = vbase[(size_t)key * DV + 16 + mr];
    }
    acc0 = wmma_f16(ap, bv0, acc0);
    acc1 = wmma_f16(ap, bv1, acc1);
  }
#pragma unroll
  for (int r = 0; r < 8; ++r) {
    int qrow = q0 + r + 8 * hf;
    float y0 = acc0[r]; y0 = y0 * fminf(fmaxf(y0 + 3.f, 0.f), 6.f) * (1.f / 6.f);
    float y1 = acc1[r]; y1 = y1 * fminf(fmaxf(y1 + 3.f, 0.f), 6.f) * (1.f / 6.f);
    Ob[((size_t)b * QPAD + qrow) * DH_ + h * DV + mr]      = (half_t)y0;
    Ob[((size_t)b * QPAD + qrow) * DH_ + h * DV + 16 + mr] = (half_t)y1;
  }
}

// ---------- K6: projection GEMM + BN, write fp32 output ----------
__global__ __launch_bounds__(256)
void proj_gemm(const half_t* __restrict__ Ob, const half_t* __restrict__ Wh,
               const float* __restrict__ s, const float* __restrict__ t,
               float* __restrict__ out) {
  int wid  = (blockIdx.x * 256 + threadIdx.x) >> 5;
  int lane = threadIdx.x & 31;
  int b = wid / (QT * 24); int rem = wid % (QT * 24);
  int qt = rem / 24, nt = rem % 24;
  int mr = lane & 15, hf = lane >> 4;
  const half_t* arow = Ob + ((size_t)b * QPAD + qt * 16 + mr) * DH_;
  const half_t* brow = Wh + (size_t)(nt * 16 + mr) * DH_;
  v8f acc = {};
  for (int k0 = 0; k0 < DH_; k0 += 32) {
    v16h a, bfr;
#pragma unroll
    for (int e = 0; e < 16; ++e) {
      a[e]   = arow[k0 + a_frag_k(e, hf)];
      bfr[e] = brow[k0 + a_frag_k(e, hf)];
    }
    acc = wmma_f16(a, bfr, acc);
  }
  int col = nt * 16 + (lane & 15);
  float sc = s[col], tc = t[col];
#pragma unroll
  for (int r = 0; r < 8; ++r) {
    int qrow = qt * 16 + r + 8 * hf;
    if (qrow < QQ)
      out[((size_t)b * QQ + qrow) * OUTC + col] = acc[r] * sc + tc;
  }
}

// ---------- K7: ga = renorm((1-t)*ga_sub + t*cls_attn) ----------
__global__ __launch_bounds__(256)
void ga_final(const float* __restrict__ ga_in, const int* __restrict__ idxbuf,
              const float* __restrict__ cls_part, const float* __restrict__ oldscale,
              const float* __restrict__ lt, float* __restrict__ out_ga) {
  int b = blockIdx.x, tid = threadIdx.x;
  float tsig = 1.f / (1.f + __expf(-lt[0]));
  float val = 0.f;
  if (tid < 196) {
    int i = tid / 14, j = tid % 14;
    int p784 = i * 56 + 2 * j;
    float gasub = ga_in[(size_t)b * PP + idxbuf[(size_t)b * PP + p784]];
    float clsv = 0.f;
#pragma unroll
    for (int h = 0; h < HH; ++h) clsv += cls_part[((size_t)b * HH + h) * 196 + tid];
    clsv *= (1.f / HH);
    val = (1.f - tsig) * gasub + tsig * clsv;
  }
  __shared__ float red[256];
  red[tid] = val; __syncthreads();
  for (int o = 128; o > 0; o >>= 1) { if (tid < o) red[tid] += red[tid + o]; __syncthreads(); }
  float scale = oldscale[b] / red[0];
  if (tid < 196) out_ga[(size_t)b * 196 + tid] = val * scale;
}

extern "C" void kernel_launch(void* const* d_in, const int* in_sizes, int n_in,
                              void* d_out, int out_size, void* d_ws, size_t ws_size,
                              hipStream_t stream) {
  const float* x      = (const float*)d_in[0];
  const float* ga     = (const float*)d_in[1];
  const float* ori    = (const float*)d_in[2];
  const float* poss   = (const float*)d_in[3];
  const float* kv_w   = (const float*)d_in[4];
  const float* kv_g   = (const float*)d_in[5];
  const float* kv_b   = (const float*)d_in[6];
  const float* kv_m   = (const float*)d_in[7];
  const float* kv_v   = (const float*)d_in[8];
  const float* q_w    = (const float*)d_in[9];
  const float* q_g    = (const float*)d_in[10];
  const float* q_b    = (const float*)d_in[11];
  const float* q_m    = (const float*)d_in[12];
  const float* q_v    = (const float*)d_in[13];
  const float* pj_w   = (const float*)d_in[14];
  const float* pj_g   = (const float*)d_in[15];
  const float* pj_b   = (const float*)d_in[16];
  const float* pj_m   = (const float*)d_in[17];
  const float* pj_v   = (const float*)d_in[18];
  const float* lt     = (const float*)d_in[19];

  char* ws = (char*)d_ws;
  size_t off = 0;
  auto alloc = [&](size_t bytes) -> void* {
    void* p = ws + off;
    off += (bytes + 255) & ~(size_t)255;
    return p;
  };
  int*    idxbuf   = (int*)   alloc((size_t)B_ * PP * 4);
  float*  oldscale = (float*) alloc((size_t)B_ * 4);
  float*  cls_part = (float*) alloc((size_t)B_ * HH * 196 * 4);
  half_t* kvw_h    = (half_t*)alloc((size_t)HTOT_ * CC * 2);
  half_t* qw_h     = (half_t*)alloc((size_t)NHKD_ * CC * 2);
  half_t* pjw_h    = (half_t*)alloc((size_t)OUTC * CC * 2);
  float*  kv_s     = (float*) alloc((size_t)HTOT_ * 4);
  float*  kv_t     = (float*) alloc((size_t)HTOT_ * 4);
  float*  q_s      = (float*) alloc((size_t)NHKD_ * 4);
  float*  q_t      = (float*) alloc((size_t)NHKD_ * 4);
  float*  pj_s     = (float*) alloc((size_t)OUTC * 4);
  float*  pj_t     = (float*) alloc((size_t)OUTC * 4);
  half_t* xh       = (half_t*)alloc((size_t)B_ * NPAD * CC * 2);
  half_t* Kb       = (half_t*)alloc((size_t)B_ * HH * NPAD * KDIM * 2);
  half_t* Vb       = (half_t*)alloc((size_t)B_ * HH * NPAD * DV * 2);
  half_t* Qb       = (half_t*)alloc((size_t)B_ * HH * QPAD * KDIM * 2);
  half_t* Ob       = (half_t*)alloc((size_t)B_ * QPAD * DH_ * 2);

  float* o_out  = (float*)d_out;
  float* ga_out = o_out + (size_t)B_ * QQ * OUTC;

  perm_oldscale<<<B_, 256, 0, stream>>>(ori, ga, idxbuf, oldscale);
  fold_weights<<<(HTOT_ * CC + 255) / 256, 256, 0, stream>>>(
      kv_w, q_w, pj_w, kv_g, kv_b, kv_m, kv_v, q_g, q_b, q_m, q_v,
      pj_g, pj_b, pj_m, pj_v, kvw_h, qw_h, pjw_h, kv_s, kv_t, q_s, q_t, pj_s, pj_t);
  gather_pos<<<B_ * NPAD, 256, 0, stream>>>(x, poss, idxbuf, xh);

  // B batches * 50 m-tiles * 24 n-tiles = 153600 wave-tiles; 8 waves/block
  kv_gemm<<<(B_ * NT50 * 24) / 8, 256, 0, stream>>>(xh, kvw_h, kv_s, kv_t, Kb, Vb);
  // B * 13 qtiles * 8 ntiles waves / 8
  q_gemm<<<B_ * QT, 256, 0, stream>>>(xh, qw_h, q_s, q_t, Qb);
  attn_kernel<<<B_ * HH * QT, 32, 0, stream>>>(Qb, Kb, Vb, cls_part, Ob);
  // B * 13 qtiles * 24 ntiles waves / 8
  proj_gemm<<<B_ * QT * 3, 256, 0, stream>>>(Ob, pjw_h, pj_s, pj_t, o_out);
  ga_final<<<B_, 256, 0, stream>>>(ga, idxbuf, cls_part, oldscale, lt, ga_out);
}
